// SelfAttention_91018946937009
// MI455X (gfx1250) — compile-verified, matrix operand order checked by reading, not run on hardware
//
#include <hip/hip_runtime.h>
#include <hip/hip_bf16.h>
#include <cstdint>

// ---------------- CDNA5 WMMA types ----------------
typedef __attribute__((ext_vector_type(16))) __bf16 v16bf;
typedef __attribute__((ext_vector_type(8)))  float  v8f;

#define S_LEN 4096
#define D_EMB 512
#define NH    8
#define DH    64

__device__ __forceinline__ uint16_t f2bf(float f) {
  uint32_t u = __float_as_uint(f);
  uint32_t r = u + 0x7FFFu + ((u >> 16) & 1u);   // round-to-nearest-even
  return (uint16_t)(r >> 16);
}

union FragU { v16bf v; uint4 q[2]; };

// Load a 16x32 bf16 fragment (A layout: lane&15 = row, lane>>4 picks K sub-block,
// element i -> K = k0 + (i&7) + 8*half + 16*(i>=8)).  Same loader serves B
// fragments when the B matrix is stored transposed (N x K row-major).
__device__ __forceinline__ v16bf load_frag(const uint16_t* base, int row, int ld,
                                           int k0, int lane) {
  const int h = (lane >> 4) & 1;
  const uint16_t* p = base + (size_t)row * (size_t)ld + k0 + 8 * h;
  FragU f;
  f.q[0] = *(const uint4*)(p);        // K = k0+8h   .. k0+8h+7
  f.q[1] = *(const uint4*)(p + 16);   // K = k0+8h+16.. k0+8h+23
  return f.v;
}

__device__ __forceinline__ v8f wmma_bf16(v16bf a, v16bf b, v8f c) {
  return __builtin_amdgcn_wmma_f32_16x16x32_bf16(false, a, false, b,
                                                 (short)0, c, false, false);
}

// ---------------- conversion kernels ----------------
__global__ void k_cvt_x(const float* __restrict__ x, uint16_t* __restrict__ xb, int n) {
  int i = blockIdx.x * blockDim.x + threadIdx.x;
  if (i < n) xb[i] = f2bf(x[i]);
}

// w: [rows][cols] fp32  ->  wt: [cols][rows] bf16
__global__ void k_transpose(const float* __restrict__ w, uint16_t* __restrict__ wt,
                            int rows, int cols) {
  int i = blockIdx.x * blockDim.x + threadIdx.x;
  if (i < rows * cols) {
    int n = i / rows;
    int k = i % rows;
    wt[i] = f2bf(w[(size_t)k * cols + n]);
  }
}

// ---------------- QKV projection GEMM ----------------
// qkv = xb[4096x512] @ w_in[512x1536] + b_in ; scatter to Q,K [H][S][64] bf16
// and V transposed [H][64][S] bf16.
__global__ __launch_bounds__(256) void k_qkv_gemm(
    const uint16_t* __restrict__ xb, const uint16_t* __restrict__ w_inT,
    const float* __restrict__ b_in,
    uint16_t* __restrict__ qb, uint16_t* __restrict__ kbuf, uint16_t* __restrict__ vT) {
  const int lane  = threadIdx.x & 31;
  const int wave  = threadIdx.x >> 5;
  const int arow0 = blockIdx.y * 256 + wave * 32;
  const int c0    = blockIdx.x * 64;

  v8f zero = {};
  v8f acc[2][4];
#pragma unroll
  for (int r = 0; r < 2; ++r)
#pragma unroll
    for (int t = 0; t < 4; ++t) acc[r][t] = zero;

  for (int k0 = 0; k0 < D_EMB; k0 += 32) {
    v16bf a0 = load_frag(xb, arow0 + (lane & 15),      D_EMB, k0, lane);
    v16bf a1 = load_frag(xb, arow0 + 16 + (lane & 15), D_EMB, k0, lane);
#pragma unroll
    for (int t = 0; t < 4; ++t) {
      v16bf b = load_frag(w_inT, c0 + t * 16 + (lane & 15), D_EMB, k0, lane);
      acc[0][t] = wmma_bf16(a0, b, acc[0][t]);
      acc[1][t] = wmma_bf16(a1, b, acc[1][t]);
    }
  }

  const int n = lane & 15, h = (lane >> 4) & 1;
#pragma unroll
  for (int r = 0; r < 2; ++r)
#pragma unroll
    for (int t = 0; t < 4; ++t) {
      int c = c0 + t * 16 + n;
      float bv = b_in[c];
#pragma unroll
      for (int e = 0; e < 8; ++e) {
        int s = arow0 + r * 16 + 8 * h + e;
        uint16_t val = f2bf(acc[r][t][e] + bv);
        if (c < 512) {
          qb[(((c >> 6) * (size_t)S_LEN) + s) * DH + (c & 63)] = val;
        } else if (c < 1024) {
          int cc = c - 512;
          kbuf[(((cc >> 6) * (size_t)S_LEN) + s) * DH + (cc & 63)] = val;
        } else {
          int cc = c - 1024;
          vT[(((cc >> 6) * (size_t)DH) + (cc & 63)) * S_LEN + s] = val;
        }
      }
    }
}

// ---------------- flash attention ----------------
// One block = one head x 128 query rows (8 waves x 16 rows).  KV blocks of 64
// are cooperatively staged into LDS once per block (8x reuse), online softmax
// per wave, P goes through a private LDS region for C->A layout conversion.
__global__ __launch_bounds__(256) void k_flash_attn(
    const uint16_t* __restrict__ qb, const uint16_t* __restrict__ kbuf,
    const uint16_t* __restrict__ vT, uint16_t* __restrict__ ab) {
  __shared__ uint16_t kTile[64 * 64];            // K rows (N x K layout), 8 KB
  __shared__ uint16_t vTile[64 * 64];            // V^T rows (N x K layout), 8 KB
  __shared__ uint16_t plds[8][16 * 64];          // per-wave P staging, 16 KB
  const int tid  = threadIdx.x;
  const int lane = tid & 31;
  const int wave = tid >> 5;
  const int head = blockIdx.x >> 5;              // 32 blocks per head
  const int q0   = (blockIdx.x & 31) * 128 + wave * 16;
  const int n = lane & 15, h = (lane >> 4) & 1;

  const uint16_t* qh = qb   + (size_t)head * S_LEN * DH;
  const uint16_t* kh = kbuf + (size_t)head * S_LEN * DH;
  const uint16_t* vh = vT   + (size_t)head * DH * S_LEN;
  uint16_t* pl = plds[wave];

  v16bf qf0 = load_frag(qh, q0 + n, DH, 0,  lane);
  v16bf qf1 = load_frag(qh, q0 + n, DH, 32, lane);

  v8f zero = {};
  v8f o[4];
#pragma unroll
  for (int t = 0; t < 4; ++t) o[t] = zero;
  float mrun[8], lrun[8];
#pragma unroll
  for (int e = 0; e < 8; ++e) { mrun[e] = -__builtin_inff(); lrun[e] = 0.0f; }

  for (int kb0 = 0; kb0 < S_LEN; kb0 += 64) {
    // ---- cooperative stage: K tile [64 kv][64 d], V^T tile [64 d][64 kv] ----
#pragma unroll
    for (int i = tid; i < 512; i += 256) {       // 512 uint4 quanta per tile
      int r = i >> 3, c = (i & 7) * 8;
      *(uint4*)(kTile + r * 64 + c) =
          *(const uint4*)(kh + (size_t)(kb0 + r) * DH + c);
      *(uint4*)(vTile + r * 64 + c) =
          *(const uint4*)(vh + (size_t)r * S_LEN + kb0 + c);
    }
    if (kb0 + 64 < S_LEN) {                      // prefetch next KV tiles
      int i = tid * 2;
      int r = i >> 3, c = (i & 7) * 8;
      __builtin_prefetch(kh + (size_t)(kb0 + 64 + r) * DH + c, 0, 1);
      __builtin_prefetch(vh + (size_t)r * S_LEN + kb0 + 64 + c, 0, 1);
    }
    __syncthreads();

    // ---- S = Q @ K^T (scaled), K fragments from LDS ----
    v8f st[4];
#pragma unroll
    for (int t = 0; t < 4; ++t) st[t] = zero;
#pragma unroll
    for (int t = 0; t < 4; ++t) {
      v16bf kf0 = load_frag(kTile, t * 16 + n, 64, 0,  lane);
      v16bf kf1 = load_frag(kTile, t * 16 + n, 64, 32, lane);
      st[t] = wmma_bf16(qf0, kf0, st[t]);
      st[t] = wmma_bf16(qf1, kf1, st[t]);
    }
#pragma unroll
    for (int t = 0; t < 4; ++t)
#pragma unroll
      for (int e = 0; e < 8; ++e) st[t][e] *= 0.125f;   // 1/sqrt(64)

    // ---- online softmax (rows live in elements; columns in 16-lane halves) ----
    float alpha[8];
#pragma unroll
    for (int e = 0; e < 8; ++e) {
      float m = fmaxf(fmaxf(st[0][e], st[1][e]), fmaxf(st[2][e], st[3][e]));
#pragma unroll
      for (int msk = 1; msk <= 8; msk <<= 1) m = fmaxf(m, __shfl_xor(m, msk));
      float mn = fmaxf(mrun[e], m);
      alpha[e] = __expf(mrun[e] - mn);
      mrun[e]  = mn;
      float s0 = __expf(st[0][e] - mn); st[0][e] = s0;
      float s1 = __expf(st[1][e] - mn); st[1][e] = s1;
      float s2 = __expf(st[2][e] - mn); st[2][e] = s2;
      float s3 = __expf(st[3][e] - mn); st[3][e] = s3;
      float l = s0 + s1 + s2 + s3;
#pragma unroll
      for (int msk = 1; msk <= 8; msk <<= 1) l += __shfl_xor(l, msk);
      lrun[e] = lrun[e] * alpha[e] + l;
    }

    // ---- P: C-layout -> private LDS -> A-layout (DS in-order within wave) ----
#pragma unroll
    for (int t = 0; t < 4; ++t)
#pragma unroll
      for (int e = 0; e < 8; ++e)
        pl[(8 * h + e) * 64 + t * 16 + n] = f2bf(st[t][e]);
#pragma unroll
    for (int t = 0; t < 4; ++t)
#pragma unroll
      for (int e = 0; e < 8; ++e) o[t][e] *= alpha[e];

    v16bf pf0 = load_frag(pl, n, 64, 0,  lane);
    v16bf pf1 = load_frag(pl, n, 64, 32, lane);

    // ---- O += P @ V, V^T fragments from LDS ----
#pragma unroll
    for (int t = 0; t < 4; ++t) {
      v16bf vf0 = load_frag(vTile, t * 16 + n, 64, 0,  lane);
      v16bf vf1 = load_frag(vTile, t * 16 + n, 64, 32, lane);
      o[t] = wmma_bf16(pf0, vf0, o[t]);
      o[t] = wmma_bf16(pf1, vf1, o[t]);
    }
    __syncthreads();                             // protect tiles before restage
  }

  // ---- epilogue: normalize, store bf16 [S][D] for out-proj ----
#pragma unroll
  for (int e = 0; e < 8; ++e) {
    float inv = 1.0f / lrun[e];
    int s = q0 + 8 * h + e;
#pragma unroll
    for (int t = 0; t < 4; ++t)
      ab[(size_t)s * D_EMB + head * DH + t * 16 + n] = f2bf(o[t][e] * inv);
  }
}

// ---------------- output projection GEMM (fp32 out + bias) ----------------
__global__ __launch_bounds__(256) void k_out_gemm(
    const uint16_t* __restrict__ ab, const uint16_t* __restrict__ w_outT,
    const float* __restrict__ b_out, float* __restrict__ out) {
  const int lane  = threadIdx.x & 31;
  const int wave  = threadIdx.x >> 5;
  const int arow0 = blockIdx.y * 256 + wave * 32;
  const int c0    = blockIdx.x * 64;

  v8f zero = {};
  v8f acc[2][4];
#pragma unroll
  for (int r = 0; r < 2; ++r)
#pragma unroll
    for (int t = 0; t < 4; ++t) acc[r][t] = zero;

  for (int k0 = 0; k0 < D_EMB; k0 += 32) {
    v16bf a0 = load_frag(ab, arow0 + (lane & 15),      D_EMB, k0, lane);
    v16bf a1 = load_frag(ab, arow0 + 16 + (lane & 15), D_EMB, k0, lane);
#pragma unroll
    for (int t = 0; t < 4; ++t) {
      v16bf b = load_frag(w_outT, c0 + t * 16 + (lane & 15), D_EMB, k0, lane);
      acc[0][t] = wmma_bf16(a0, b, acc[0][t]);
      acc[1][t] = wmma_bf16(a1, b, acc[1][t]);
    }
  }

  const int n = lane & 15, h = (lane >> 4) & 1;
#pragma unroll
  for (int r = 0; r < 2; ++r)
#pragma unroll
    for (int t = 0; t < 4; ++t) {
      int c = c0 + t * 16 + n;
      float bv = b_out[c];
#pragma unroll
      for (int e = 0; e < 8; ++e) {
        int s = arow0 + r * 16 + 8 * h + e;
        out[(size_t)s * D_EMB + c] = acc[r][t][e] + bv;
      }
    }
}

// ---------------- launcher ----------------
extern "C" void kernel_launch(void* const* d_in, const int* in_sizes, int n_in,
                              void* d_out, int out_size, void* d_ws, size_t ws_size,
                              hipStream_t stream) {
  (void)in_sizes; (void)n_in; (void)out_size; (void)ws_size;
  const float* x     = (const float*)d_in[0];
  const float* w_in  = (const float*)d_in[1];
  const float* b_in  = (const float*)d_in[2];
  const float* w_out = (const float*)d_in[3];
  const float* b_out = (const float*)d_in[4];
  float* out = (float*)d_out;

  char* ws = (char*)d_ws;
  uint16_t* xb     = (uint16_t*)ws; ws += (size_t)S_LEN * D_EMB * 2;      // 4 MB
  uint16_t* w_inT  = (uint16_t*)ws; ws += (size_t)3 * D_EMB * D_EMB * 2;  // 1.5 MB
  uint16_t* w_outT = (uint16_t*)ws; ws += (size_t)D_EMB * D_EMB * 2;      // 0.5 MB
  uint16_t* qb     = (uint16_t*)ws; ws += (size_t)NH * S_LEN * DH * 2;    // 4 MB
  uint16_t* kb     = (uint16_t*)ws; ws += (size_t)NH * S_LEN * DH * 2;    // 4 MB
  uint16_t* vT     = (uint16_t*)ws; ws += (size_t)NH * DH * S_LEN * 2;    // 4 MB
  uint16_t* ab     = (uint16_t*)ws;                                       // 4 MB

  k_cvt_x    <<<(S_LEN * D_EMB) / 256, 256, 0, stream>>>(x, xb, S_LEN * D_EMB);
  k_transpose<<<(3 * D_EMB * D_EMB) / 256, 256, 0, stream>>>(w_in,  w_inT,  D_EMB, 3 * D_EMB);
  k_transpose<<<(D_EMB * D_EMB) / 256, 256, 0, stream>>>(w_out, w_outT, D_EMB, D_EMB);

  k_qkv_gemm  <<<dim3(24, 16), 256, 0, stream>>>(xb, w_inT, b_in, qb, kb, vT);
  k_flash_attn<<<256,          256, 0, stream>>>(qb, kb, vT, ab);
  k_out_gemm  <<<dim3(8, 16),  256, 0, stream>>>(ab, w_outT, b_out, out);
}
